// PointNet2_90993177133213
// MI455X (gfx1250) — matrix operand order repaired and checked by reading, not compiled, not run on hardware
//
#include <hip/hip_runtime.h>
#include <stdint.h>

// ---------------------------------------------------------------------------
// Types for CDNA5 WMMA (wave32)
// ---------------------------------------------------------------------------
typedef __attribute__((ext_vector_type(16))) __bf16          v16bf;
typedef __attribute__((ext_vector_type(16))) unsigned short  v16us;
typedef __attribute__((ext_vector_type(8)))  unsigned short  v8us;
typedef __attribute__((ext_vector_type(8)))  float           v8f;

__device__ __forceinline__ unsigned short f2bf(float x) {
    unsigned u = __float_as_uint(x);
    u += 0x7fffu + ((u >> 16) & 1u);      // round-to-nearest-even
    return (unsigned short)(u >> 16);
}

__device__ __forceinline__ v16us cat8(v8us lo, v8us hi) {
    return __builtin_shufflevector(lo, hi, 0, 1, 2, 3, 4, 5, 6, 7,
                                           8, 9, 10, 11, 12, 13, 14, 15);
}

__device__ __forceinline__ v8f wmma_bf16(v16us a, v16us b, v8f c) {
    return __builtin_amdgcn_wmma_f32_16x16x32_bf16(
        false, __builtin_bit_cast(v16bf, a),
        false, __builtin_bit_cast(v16bf, b),
        (short)0, c, false, false);
}

// ---------------------------------------------------------------------------
// Farthest point sampling: one block per batch. dist lives in registers
// (N/256 <= 16 per thread), argmax tree-reduce in LDS with first-index
// tie-break to match jnp.argmax.
// ---------------------------------------------------------------------------
__global__ __launch_bounds__(256)
void fps_kernel(const float* __restrict__ xyz, int N, int S,
                int* __restrict__ idx_out, float* __restrict__ new_xyz)
{
    const int b = blockIdx.x;
    const float* p = xyz + (size_t)b * N * 3;
    const int t = threadIdx.x, T = 256;

    float dist[16];
    #pragma unroll
    for (int j = 0; j < 16; ++j) dist[j] = 1e10f;
    const int npt = (N + T - 1) / T;

    __shared__ float smax[256];
    __shared__ int   simax[256];
    __shared__ float c0, c1, c2;

    int far = 0;
    for (int i = 0; i < S; ++i) {
        if (t == 0) {
            float cx = p[far * 3 + 0], cy = p[far * 3 + 1], cz = p[far * 3 + 2];
            idx_out[b * S + i] = far;
            float* nx = new_xyz + ((size_t)b * S + i) * 3;
            nx[0] = cx; nx[1] = cy; nx[2] = cz;
            c0 = cx; c1 = cy; c2 = cz;
        }
        __syncthreads();
        const float cx = c0, cy = c1, cz = c2;

        float best = -1.0f; int bi = 0;
        for (int j = 0; j < npt; ++j) {
            const int pi = t + j * T;
            if (pi < N) {
                float dx = p[pi * 3 + 0] - cx;
                float dy = p[pi * 3 + 1] - cy;
                float dz = p[pi * 3 + 2] - cz;
                float d  = fminf(dist[j], dx * dx + dy * dy + dz * dz);
                dist[j] = d;
                if (d > best) { best = d; bi = pi; }   // ascending pi -> first max kept
            }
        }
        smax[t] = best; simax[t] = bi;
        __syncthreads();
        for (int off = T / 2; off > 0; off >>= 1) {
            if (t < off) {
                float ov = smax[t + off]; int oi = simax[t + off];
                if (ov > smax[t] || (ov == smax[t] && oi < simax[t])) {
                    smax[t] = ov; simax[t] = oi;
                }
            }
            __syncthreads();
        }
        far = simax[0];
        __syncthreads();
    }
}

// ---------------------------------------------------------------------------
// Ball query: first `ns` indices (ascending) with d^2 <= r^2, padded with
// the first hit. One block per batch, queries strided over threads.
// ---------------------------------------------------------------------------
__global__ __launch_bounds__(256)
void ball_query_kernel(const float* __restrict__ xyz, const float* __restrict__ q,
                       int N, int S, float r2, int ns, int* __restrict__ gidx)
{
    const int b = blockIdx.x;
    const float* p = xyz + (size_t)b * N * 3;
    for (int s = threadIdx.x; s < S; s += blockDim.x) {
        const float* c = q + ((size_t)b * S + s) * 3;
        const float cx = c[0], cy = c[1], cz = c[2];
        int* out = gidx + ((size_t)b * S + s) * ns;
        int cnt = 0, first = 0; bool have = false;
        for (int j = 0; j < N && cnt < ns; ++j) {
            float dx = p[j * 3 + 0] - cx;
            float dy = p[j * 3 + 1] - cy;
            float dz = p[j * 3 + 2] - cz;
            if (dx * dx + dy * dy + dz * dz <= r2) {
                if (!have) { first = j; have = true; }
                out[cnt++] = j;
            }
        }
        for (; cnt < ns; ++cnt) out[cnt] = first;
    }
}

// ---------------------------------------------------------------------------
// Weight f32 -> bf16 with zero K-padding (W stored (cout, cinPad) row-major)
// ---------------------------------------------------------------------------
__global__ __launch_bounds__(256)
void wconv_kernel(const float* __restrict__ w, unsigned short* __restrict__ o,
                  int cout, int cin, int cinPad)
{
    const int i = blockIdx.x * blockDim.x + threadIdx.x;
    if (i >= cout * cinPad) return;
    const int co = i / cinPad, ci = i % cinPad;
    o[i] = (ci < cin) ? f2bf(w[(size_t)co * cin + ci]) : (unsigned short)0;
}

// ---------------------------------------------------------------------------
// Grouping: build bf16 activation rows (one thread per row)
// ---------------------------------------------------------------------------
__global__ __launch_bounds__(256)
void group_sa1_kernel(const float* __restrict__ xyz, const float* __restrict__ nxyz,
                      const int* __restrict__ gidx, unsigned short* __restrict__ A,
                      int N, int S, int ns)
{
    const int row = blockIdx.x * blockDim.x + threadIdx.x;
    const int b = row / (S * ns);
    const int s = (row % (S * ns)) / ns;
    const int j = gidx[row];
    const float* p = xyz  + ((size_t)b * N + j) * 3;
    const float* c = nxyz + ((size_t)b * S + s) * 3;
    unsigned short* o = A + (size_t)row * 32;
    o[0] = f2bf(p[0] - c[0]);
    o[1] = f2bf(p[1] - c[1]);
    o[2] = f2bf(p[2] - c[2]);
    #pragma unroll
    for (int k = 3; k < 32; ++k) o[k] = 0;
}

__global__ __launch_bounds__(256)
void group_sa2_kernel(const float* __restrict__ xyz, const float* __restrict__ nxyz,
                      const float* __restrict__ feat, const int* __restrict__ gidx,
                      unsigned short* __restrict__ A, int N, int S, int ns)
{
    const int row = blockIdx.x * blockDim.x + threadIdx.x;
    const int b = row / (S * ns);
    const int s = (row % (S * ns)) / ns;
    const int j = gidx[row];
    const float* p = xyz  + ((size_t)b * N + j) * 3;
    const float* c = nxyz + ((size_t)b * S + s) * 3;
    const float* f = feat + ((size_t)b * N + j) * 128;
    unsigned short* o = A + (size_t)row * 160;
    o[0] = f2bf(p[0] - c[0]);
    o[1] = f2bf(p[1] - c[1]);
    o[2] = f2bf(p[2] - c[2]);
    for (int k = 0; k < 128; ++k) o[3 + k] = f2bf(f[k]);
    #pragma unroll
    for (int k = 131; k < 160; ++k) o[k] = 0;
}

__global__ __launch_bounds__(256)
void group_sa3_kernel(const float* __restrict__ xyz, const float* __restrict__ feat,
                      unsigned short* __restrict__ A, int S)
{
    const int row = blockIdx.x * blockDim.x + threadIdx.x;   // b*S + s
    const float* p = xyz  + (size_t)row * 3;
    const float* f = feat + (size_t)row * 256;
    unsigned short* o = A + (size_t)row * 288;
    o[0] = f2bf(p[0]); o[1] = f2bf(p[1]); o[2] = f2bf(p[2]);
    for (int k = 0; k < 256; ++k) o[3 + k] = f2bf(f[k]);
    #pragma unroll
    for (int k = 259; k < 288; ++k) o[k] = 0;
}

// ---------------------------------------------------------------------------
// Tiled bf16 WMMA GEMM:  Y = relu(A[M,K] * W[N,K]^T + bias)
// Block = 8 waves (4x2), macro tile 128x64, BK = 32.
// Each wave computes a 32x32 patch = 2x2 WMMA tiles (4 wmma / K-step,
// A- and B-fragment reuse). Double-buffered LDS staging: global->VGPR
// prefetch of slice k+1 overlaps WMMA on slice k; ONE barrier per K-step.
// If pool != nullptr: fused max-pool over 2^nsShift rows via atomicMax on
// the int view (valid: post-ReLU >= 0, pool pre-zeroed). Else bf16 to Cout.
// M % 128 == 0, N % 64 == 0, K % 32 == 0 guaranteed by caller.
// ---------------------------------------------------------------------------
__global__ __launch_bounds__(256)
void gemm_bf16_wmma_kernel(const unsigned short* __restrict__ A,
                           const unsigned short* __restrict__ W,
                           const float* __restrict__ bias,
                           unsigned short* __restrict__ Cout,
                           float* __restrict__ pool,
                           int M, int K, int N, int nsShift)
{
    __shared__ __align__(16) unsigned short As[2][128 * 32];  // 2 x 8 KB
    __shared__ __align__(16) unsigned short Ws[2][64 * 32];   // 2 x 4 KB

    const int t    = threadIdx.x;
    const int wave = t >> 5;
    const int lane = t & 31;
    const int half = lane >> 4;      // 0: lanes 0-15, 1: lanes 16-31
    const int r    = lane & 15;
    const int wm   = wave >> 1;      // 0..3 : 32-row slice
    const int wn   = wave & 1;       // 0..1 : 32-col slice
    const int mTile = blockIdx.x * 128;
    const int nTile = blockIdx.y * 64;

    v8f acc00 = {}, acc01 = {}, acc10 = {}, acc11 = {};

    // cooperative staging mapping: A 128x32 (16 us/thread), W 64x32 (8 us/thread)
    const int arow = t >> 1, acol = (t & 1) * 16;
    const int wrow = t >> 2, wcol = (t & 3) * 8;
    const unsigned short* Ag = A + (size_t)(mTile + arow) * K + acol;
    const unsigned short* Wg = W + (size_t)(nTile + wrow) * K + wcol;

    // LDS fragment bases (relative to buffer)
    const int am0 = (wm * 32 + r) * 32;
    const int am1 = (wm * 32 + 16 + r) * 32;
    const int bn0 = (wn * 32 + r) * 32;
    const int bn1 = (wn * 32 + 16 + r) * 32;

    // prologue: stage slice 0
    v8us ra0 = *(const v8us*)(Ag);
    v8us ra1 = *(const v8us*)(Ag + 8);
    v8us rw  = *(const v8us*)(Wg);
    *(v8us*)&As[0][arow * 32 + acol]     = ra0;
    *(v8us*)&As[0][arow * 32 + acol + 8] = ra1;
    *(v8us*)&Ws[0][wrow * 32 + wcol]     = rw;
    __syncthreads();

    int cur = 0;
    for (int k0 = 0; k0 < K; k0 += 32) {
        const bool more = (k0 + 32) < K;
        if (more) {                       // prefetch next slice into VGPRs
            ra0 = *(const v8us*)(Ag + k0 + 32);
            ra1 = *(const v8us*)(Ag + k0 + 40);
            rw  = *(const v8us*)(Wg + k0 + 32);
            if (k0 + 64 < K) {            // L2 prefetch for the slice after
                __builtin_prefetch(Ag + k0 + 64, 0, 1);
                __builtin_prefetch(Wg + k0 + 64, 0, 1);
            }
        }

        const unsigned short* as = As[cur];
        const unsigned short* ws = Ws[cur];
        // A fragments: lanes0-15 K 0-7|16-23, lanes16-31 K 8-15|24-31
        v16us a0 = cat8(*(const v8us*)&as[am0 + half * 8],
                        *(const v8us*)&as[am0 + 16 + half * 8]);
        v16us a1 = cat8(*(const v8us*)&as[am1 + half * 8],
                        *(const v8us*)&as[am1 + 16 + half * 8]);
        // B fragments: lanes0-15 K 0-7|16-23, lanes16-31 K 8-15|24-31
        v16us b0 = cat8(*(const v8us*)&ws[bn0 + half * 8],
                        *(const v8us*)&ws[bn0 + 16 + half * 8]);
        v16us b1 = cat8(*(const v8us*)&ws[bn1 + half * 8],
                        *(const v8us*)&ws[bn1 + 16 + half * 8]);

        acc00 = wmma_bf16(a0, b0, acc00);
        acc01 = wmma_bf16(a0, b1, acc01);
        acc10 = wmma_bf16(a1, b0, acc10);
        acc11 = wmma_bf16(a1, b1, acc11);

        if (more) {
            // safe: buffer cur^1 was last read before the previous barrier
            *(v8us*)&As[cur ^ 1][arow * 32 + acol]     = ra0;
            *(v8us*)&As[cur ^ 1][arow * 32 + acol + 8] = ra1;
            *(v8us*)&Ws[cur ^ 1][wrow * 32 + wcol]     = rw;
            __syncthreads();
            cur ^= 1;
        }
    }

    // Epilogue: C layout — lane col = r, VGPR j row = half*8 + j
    auto emit = [&](v8f acc, int subm, int subn) {
        const int colg = nTile + wn * 32 + subn * 16 + r;
        const float bv = bias[colg];
        const int rowb = mTile + wm * 32 + subm * 16 + half * 8;
        if (pool) {
            #pragma unroll
            for (int j = 0; j < 8; ++j) {
                float v = fmaxf(acc[j] + bv, 0.0f);
                int g = (rowb + j) >> nsShift;        // nsample is a power of two
                atomicMax((int*)&pool[(size_t)g * N + colg], __float_as_int(v));
            }
        } else {
            #pragma unroll
            for (int j = 0; j < 8; ++j) {
                float v = fmaxf(acc[j] + bv, 0.0f);
                Cout[(size_t)(rowb + j) * N + colg] = f2bf(v);
            }
        }
    };
    emit(acc00, 0, 0); emit(acc01, 0, 1);
    emit(acc10, 1, 0); emit(acc11, 1, 1);
}

// ---------------------------------------------------------------------------
// Host orchestration
// ---------------------------------------------------------------------------
extern "C" void kernel_launch(void* const* d_in, const int* in_sizes, int n_in,
                              void* d_out, int out_size, void* d_ws, size_t ws_size,
                              hipStream_t stream)
{
    (void)in_sizes; (void)n_in; (void)out_size; (void)ws_size;

    const float* x   = (const float*)d_in[0];
    const float* w1a = (const float*)d_in[1];  const float* b1a = (const float*)d_in[2];
    const float* w1b = (const float*)d_in[3];  const float* b1b = (const float*)d_in[4];
    const float* w1c = (const float*)d_in[5];  const float* b1c = (const float*)d_in[6];
    const float* w2a = (const float*)d_in[7];  const float* b2a = (const float*)d_in[8];
    const float* w2b = (const float*)d_in[9];  const float* b2b = (const float*)d_in[10];
    const float* w2c = (const float*)d_in[11]; const float* b2c = (const float*)d_in[12];
    const float* w3a = (const float*)d_in[13]; const float* b3a = (const float*)d_in[14];
    const float* w3b = (const float*)d_in[15]; const float* b3b = (const float*)d_in[16];
    const float* w3c = (const float*)d_in[17]; const float* b3c = (const float*)d_in[18];

    const int B = 16, N1 = 4096, S1 = 512, NS1 = 32;
    const int S2 = 128, NS2 = 64;

    char* base = (char*)d_ws;
    size_t off = 0;
    auto carve = [&](size_t bytes) -> void* {
        void* p = base + off;
        off = (off + bytes + 255) & ~(size_t)255;
        return p;
    };

    int*   idx1  = (int*)  carve((size_t)B * S1 * 4);
    float* l1xyz = (float*)carve((size_t)B * S1 * 3 * 4);
    int*   gidx1 = (int*)  carve((size_t)B * S1 * NS1 * 4);
    int*   idx2  = (int*)  carve((size_t)B * S2 * 4);
    float* l2xyz = (float*)carve((size_t)B * S2 * 3 * 4);
    int*   gidx2 = (int*)  carve((size_t)B * S2 * NS2 * 4);
    float* l1pts = (float*)carve((size_t)B * S1 * 128 * 4);
    float* l2pts = (float*)carve((size_t)B * S2 * 256 * 4);

    unsigned short* wb1a = (unsigned short*)carve((size_t)64 * 32 * 2);
    unsigned short* wb1b = (unsigned short*)carve((size_t)64 * 64 * 2);
    unsigned short* wb1c = (unsigned short*)carve((size_t)128 * 64 * 2);
    unsigned short* wb2a = (unsigned short*)carve((size_t)128 * 160 * 2);
    unsigned short* wb2b = (unsigned short*)carve((size_t)128 * 128 * 2);
    unsigned short* wb2c = (unsigned short*)carve((size_t)256 * 128 * 2);
    unsigned short* wb3a = (unsigned short*)carve((size_t)256 * 288 * 2);
    unsigned short* wb3b = (unsigned short*)carve((size_t)512 * 256 * 2);
    unsigned short* wb3c = (unsigned short*)carve((size_t)1024 * 512 * 2);

    unsigned short* bufA = (unsigned short*)carve((size_t)131072 * 160 * 2); // 42 MB
    unsigned short* bufB = (unsigned short*)carve((size_t)262144 * 64  * 2); // 33.5 MB

    auto wc = [&](const float* w, unsigned short* o, int co, int ci, int cip) {
        int n = co * cip;
        wconv_kernel<<<(n + 255) / 256, 256, 0, stream>>>(w, o, co, ci, cip);
    };
    wc(w1a, wb1a, 64, 3, 32);     wc(w1b, wb1b, 64, 64, 64);    wc(w1c, wb1c, 128, 64, 64);
    wc(w2a, wb2a, 128, 131, 160); wc(w2b, wb2b, 128, 128, 128); wc(w2c, wb2c, 256, 128, 128);
    wc(w3a, wb3a, 256, 259, 288); wc(w3b, wb3b, 512, 256, 256); wc(w3c, wb3c, 1024, 512, 512);

    hipMemsetAsync(l1pts, 0, (size_t)B * S1 * 128 * 4, stream);
    hipMemsetAsync(l2pts, 0, (size_t)B * S2 * 256 * 4, stream);
    hipMemsetAsync(d_out, 0, (size_t)B * 1024 * 4, stream);

    auto gemm = [&](const unsigned short* A, const unsigned short* W, const float* bias,
                    unsigned short* Cout, float* pool, int M, int K, int N, int nsShift) {
        gemm_bf16_wmma_kernel<<<dim3(M / 128, N / 64), 256, 0, stream>>>(
            A, W, bias, Cout, pool, M, K, N, nsShift);
    };

    // ---- SA1: 4096 -> 512 centroids, 32 neighbors, MLP 3->64->64->128 ----
    fps_kernel<<<B, 256, 0, stream>>>(x, N1, S1, idx1, l1xyz);
    ball_query_kernel<<<B, 256, 0, stream>>>(x, l1xyz, N1, S1, 0.03f * 0.03f, NS1, gidx1);
    {
        const int rows = B * S1 * NS1;                       // 262144
        group_sa1_kernel<<<rows / 256, 256, 0, stream>>>(x, l1xyz, gidx1, bufA, N1, S1, NS1);
        gemm(bufA, wb1a, b1a, bufB, nullptr, rows, 32, 64, 0);
        gemm(bufB, wb1b, b1b, bufA, nullptr, rows, 64, 64, 0);
        gemm(bufA, wb1c, b1c, nullptr, l1pts, rows, 64, 128, 5);     // pool over 32
    }

    // ---- SA2: 512 -> 128 centroids, 64 neighbors, MLP 131->128->128->256 ----
    fps_kernel<<<B, 256, 0, stream>>>(l1xyz, S1, S2, idx2, l2xyz);
    ball_query_kernel<<<B, 256, 0, stream>>>(l1xyz, l2xyz, S1, S2, 0.06f * 0.06f, NS2, gidx2);
    {
        const int rows = B * S2 * NS2;                       // 131072
        group_sa2_kernel<<<rows / 256, 256, 0, stream>>>(l1xyz, l2xyz, l1pts, gidx2,
                                                         bufA, S1, S2, NS2);
        gemm(bufA, wb2a, b2a, bufB, nullptr, rows, 160, 128, 0);
        gemm(bufB, wb2b, b2b, bufA, nullptr, rows, 128, 128, 0);
        gemm(bufA, wb2c, b2c, nullptr, l2pts, rows, 128, 256, 6);    // pool over 64
    }

    // ---- SA3 (group-all): MLP 259->256->512->1024, max over 128 points ----
    {
        const int rows = B * S2;                             // 2048
        group_sa3_kernel<<<rows / 256, 256, 0, stream>>>(l2xyz, l2pts, bufA, S2);
        gemm(bufA, wb3a, b3a, bufB, nullptr, rows, 288, 256, 0);
        gemm(bufB, wb3b, b3b, bufA, nullptr, rows, 256, 512, 0);
        gemm(bufA, wb3c, b3c, nullptr, (float*)d_out, rows, 512, 1024, 7); // pool over 128
    }
}
